// ONOBlock_90744069030606
// MI455X (gfx1250) — matrix operand order repaired and checked by reading, not compiled, not run on hardware
//
#include <hip/hip_runtime.h>

#define BB 4
#define NN 8192
#define CC 256
#define HH 8
#define DD 32

typedef __bf16 bf16_t;
typedef __attribute__((ext_vector_type(16))) __bf16 v16bf;
typedef __attribute__((ext_vector_type(8)))  __bf16 v8bf;
typedef __attribute__((ext_vector_type(8)))  float  v8f;

union V16 { v16bf v; v8bf h[2]; };

// ---------------------------------------------------------------- utilities
__device__ __forceinline__ float blockSum256(float v, float* lds) {
#pragma unroll
  for (int m = 16; m >= 1; m >>= 1) v += __shfl_xor(v, m, 32);
  int w = threadIdx.x >> 5;
  if ((threadIdx.x & 31) == 0) lds[w] = v;
  __syncthreads();
  float s = 0.f;
#pragma unroll
  for (int i = 0; i < 8; ++i) s += lds[i];
  __syncthreads();
  return s;
}

__global__ void zero_kernel(float* __restrict__ p, int n) {
  int i = blockIdx.x * 256 + threadIdx.x;
  if (i < n) p[i] = 0.f;
}

// fp32 [b][R][Ccols] -> bf16 [b][Ccols][R]
__global__ __launch_bounds__(256) void transpose_bf_kernel(
    const float* __restrict__ src, bf16_t* __restrict__ dst, int R, int Ccols) {
  __shared__ float tile[32][33];
  int b = blockIdx.z;
  const float* s = src + (size_t)b * R * Ccols;
  bf16_t* d = dst + (size_t)b * R * Ccols;
  int r0 = blockIdx.x * 32, c0 = blockIdx.y * 32;
  int tx = threadIdx.x, ty = threadIdx.y;  // (32, 8)
#pragma unroll
  for (int j = 0; j < 32; j += 8)
    tile[ty + j][tx] = s[(size_t)(r0 + ty + j) * Ccols + c0 + tx];
  __syncthreads();
#pragma unroll
  for (int j = 0; j < 32; j += 8)
    d[(size_t)(c0 + ty + j) * R + r0 + tx] = (bf16_t)tile[tx][ty + j];
}

// -------------------------------------------------- PE + LN1/LN2 (fused)
__global__ __launch_bounds__(256) void prep_kernel(
    const float* __restrict__ input, const float* __restrict__ inF,
    const float* __restrict__ ln1g, const float* __restrict__ ln1b,
    const float* __restrict__ ln2g, const float* __restrict__ ln2b,
    bf16_t* __restrict__ xn, bf16_t* __restrict__ fn) {
  __shared__ float lds[8];
  int row = blockIdx.x;        // b*N + n
  int n = row & (NN - 1);
  int c = threadIdx.x;
  int i = c >> 1;
  float div = expf(-(float)(2 * i) * (9.210340371976184f / (float)CC)); // ln(1e4)
  float ang = (float)n * div;
  float pe = (c & 1) ? cosf(ang) : sinf(ang);
  size_t idx = (size_t)row * CC + c;
  float xv = input[idx] + pe;
  float fv = inF[idx] + pe;
  // LN1(x)
  float m = blockSum256(xv, lds) * (1.f / CC);
  float dx = xv - m;
  float v = blockSum256(dx * dx, lds) * (1.f / CC);
  xn[idx] = (bf16_t)(dx * rsqrtf(v + 1e-6f) * ln1g[c] + ln1b[c]);
  // LN2(f)
  m = blockSum256(fv, lds) * (1.f / CC);
  float df = fv - m;
  v = blockSum256(df * df, lds) * (1.f / CC);
  fn[idx] = (bf16_t)(df * rsqrtf(v + 1e-6f) * ln2g[c] + ln2b[c]);
}

// ------------------------------------------------ WMMA GEMM, Nc = 256 fixed
// KT is the (compile-time) reduction depth so all 16 B-tile addresses fold
// into one base pointer + 24-bit immediate offsets (max 15*16*KT*2 bytes).
// EPI: 0=bias->bf16  1=head-softmax->bf16  2=bias+resid->f32 & LN3->bf16
//      3=bias+GELU->bf16  4=bias+resid->f32 & x*softplus(mu)->bf16
//      5=transposed bf16 store  6=LeakyReLU->f32
template <int EPI, int KT>
__global__ __launch_bounds__(256) void gemm256(
    const bf16_t* __restrict__ A, const bf16_t* __restrict__ Wt,
    const float* __restrict__ bias, const float* __restrict__ resid,
    const float* __restrict__ aux1, const float* __restrict__ aux2,
    float* __restrict__ outF, bf16_t* __restrict__ outB,
    int M, long sA, long sW, long sOF, long sOB) {
  int bidx = blockIdx.y;
  A += (size_t)bidx * sA;
  Wt += (size_t)bidx * sW;
  if (outF) outF += (size_t)bidx * sOF;
  if (outB) outB += (size_t)bidx * sOB;

  int wave = threadIdx.x >> 5, lane = threadIdx.x & 31;
  int hi = lane >> 4;           // 0: lanes 0-15, 1: lanes 16-31
  int l16 = lane & 15;
  int mbase = blockIdx.x * 128 + wave * 16;

  v8f acc[16];
#pragma unroll
  for (int i = 0; i < 16; ++i)
#pragma unroll
    for (int j = 0; j < 8; ++j) acc[i][j] = 0.f;

  // A as loaded (contiguous K order, same as B): lane holds row l%16; low lanes K {0..15},
  // high lanes K {16..31}. B layout: lane holds col l%16, contiguous
  // 16 K values at kb + 16*hi (Wt is [n][k]).
  const bf16_t* aptr = A + (size_t)(mbase + l16) * KT + hi * 16;
  const bf16_t* wptr = Wt + (size_t)l16 * KT + hi * 16;
  for (int kb = 0; kb < KT; kb += 32) {
    V16 av;
    av.h[0] = *(const v8bf*)(aptr);
    av.h[1] = *(const v8bf*)(aptr + 8);
#pragma unroll
    for (int nt = 0; nt < 16; ++nt) {
      v16bf bv = *(const v16bf*)(wptr + nt * 16 * KT);  // immediate offset
      acc[nt] = __builtin_amdgcn_wmma_f32_16x16x32_bf16(
          false, av.v, false, bv, (short)0, acc[nt], false, false);
    }
    aptr += 32;
    wptr += 32;
  }

  // D layout: VGPR r -> row mbase + r + 8*hi, col nt*16 + l16
  int rowTop = mbase + hi * 8;

  if (EPI == 0) {
#pragma unroll
    for (int nt = 0; nt < 16; ++nt) {
      int col = nt * 16 + l16;
      float bb = bias ? bias[col] : 0.f;
#pragma unroll
      for (int r = 0; r < 8; ++r)
        outB[(size_t)(rowTop + r) * CC + col] = (bf16_t)(acc[nt][r] + bb);
    }
  } else if (EPI == 1) {
#pragma unroll
    for (int r = 0; r < 8; ++r) {
      size_t rb = (size_t)(rowTop + r) * CC;
#pragma unroll
      for (int h = 0; h < HH; ++h) {
        int c0 = h * 32 + l16, c1 = c0 + 16;
        float a0 = acc[2 * h][r] + (bias ? bias[c0] : 0.f);
        float a1 = acc[2 * h + 1][r] + (bias ? bias[c1] : 0.f);
        float mx = fmaxf(a0, a1);
#pragma unroll
        for (int msk = 8; msk >= 1; msk >>= 1) mx = fmaxf(mx, __shfl_xor(mx, msk, 32));
        float e0 = expf(a0 - mx), e1 = expf(a1 - mx);
        float s = e0 + e1;
#pragma unroll
        for (int msk = 8; msk >= 1; msk >>= 1) s += __shfl_xor(s, msk, 32);
        float inv = 1.f / s;
        outB[rb + c0] = (bf16_t)(e0 * inv);
        outB[rb + c1] = (bf16_t)(e1 * inv);
      }
    }
  } else if (EPI == 2) {
#pragma unroll
    for (int r = 0; r < 8; ++r) {
      size_t rb = (size_t)(rowTop + r) * CC;
      float xv[16], sum = 0.f;
#pragma unroll
      for (int nt = 0; nt < 16; ++nt) {
        int col = nt * 16 + l16;
        float x = acc[nt][r] + bias[col] + resid[rb + col];
        xv[nt] = x; sum += x;
      }
#pragma unroll
      for (int msk = 8; msk >= 1; msk >>= 1) sum += __shfl_xor(sum, msk, 32);
      float mean = sum * (1.f / CC), vs = 0.f;
#pragma unroll
      for (int nt = 0; nt < 16; ++nt) { float d = xv[nt] - mean; vs += d * d; }
#pragma unroll
      for (int msk = 8; msk >= 1; msk >>= 1) vs += __shfl_xor(vs, msk, 32);
      float rstd = rsqrtf(vs * (1.f / CC) + 1e-6f);
#pragma unroll
      for (int nt = 0; nt < 16; ++nt) {
        int col = nt * 16 + l16;
        outF[rb + col] = xv[nt];
        outB[rb + col] = (bf16_t)((xv[nt] - mean) * rstd * aux1[col] + aux2[col]);
      }
    }
  } else if (EPI == 3) {
#pragma unroll
    for (int nt = 0; nt < 16; ++nt) {
      int col = nt * 16 + l16;
      float bb = bias[col];
#pragma unroll
      for (int r = 0; r < 8; ++r) {
        float x = acc[nt][r] + bb;
        outB[(size_t)(rowTop + r) * CC + col] =
            (bf16_t)(0.5f * x * (1.f + erff(x * 0.70710678118f)));
      }
    }
  } else if (EPI == 4) {
#pragma unroll
    for (int nt = 0; nt < 16; ++nt) {
      int col = nt * 16 + l16;
      float bb = bias[col];
      float sp = log1pf(expf(aux1[col]));  // softplus(mu[col])
#pragma unroll
      for (int r = 0; r < 8; ++r) {
        size_t idx = (size_t)(rowTop + r) * CC + col;
        float x = acc[nt][r] + bb + resid[idx];
        outF[idx] = x;
        outB[idx] = (bf16_t)(x * sp);
      }
    }
  } else if (EPI == 5) {
#pragma unroll
    for (int nt = 0; nt < 16; ++nt) {
      int col = nt * 16 + l16;
#pragma unroll
      for (int r = 0; r < 8; ++r)
        outB[(size_t)col * M + (rowTop + r)] = (bf16_t)acc[nt][r];
    }
  } else {  // EPI == 6
#pragma unroll
    for (int nt = 0; nt < 16; ++nt) {
      int col = nt * 16 + l16;
#pragma unroll
      for (int r = 0; r < 8; ++r) {
        float x = acc[nt][r];
        outF[(size_t)(rowTop + r) * CC + col] = (x >= 0.f) ? x : 0.01f * x;
      }
    }
  }
}

// ------------------------------------- per-(b,h) kv = k^T v, ksum = sum(k)
__global__ __launch_bounds__(256) void kv_kernel(
    const bf16_t* __restrict__ kbuf, const bf16_t* __restrict__ vbuf,
    float* __restrict__ kv, float* __restrict__ ksum, int s) {
  __shared__ float ks[8][32], vs[8][32];
  int bh = blockIdx.y, b = bh / HH, h = bh % HH;
  int n0 = blockIdx.x * 256;
  int t = threadIdx.x;
  int e = t & 31, dg = t >> 5;   // each thread: fixed e, d in {dg, dg+8, dg+16, dg+24}
  float acc[4] = {0.f, 0.f, 0.f, 0.f};
  float ksacc = 0.f;
  for (int it = 0; it < 32; ++it) {
    int n = n0 + it * 8 + (t >> 5);
    size_t base = ((size_t)(b * NN + n)) * CC + h * DD + (t & 31);
    ks[t >> 5][t & 31] = (float)kbuf[base];
    vs[t >> 5][t & 31] = (float)vbuf[base];
    __syncthreads();
#pragma unroll
    for (int j = 0; j < 8; ++j) {
      float vv = vs[j][e];
#pragma unroll
      for (int p = 0; p < 4; ++p) acc[p] += ks[j][dg + 8 * p] * vv;
    }
    if (t < 32)
#pragma unroll
      for (int j = 0; j < 8; ++j) ksacc += ks[j][t];
    __syncthreads();
  }
  size_t kvbase = ((size_t)bh * 2 + s) * 1024;
#pragma unroll
  for (int p = 0; p < 4; ++p)
    atomicAdd(&kv[kvbase + (size_t)(dg + 8 * p) * 32 + e], acc[p]);
  if (t < 32) atomicAdd(&ksum[((size_t)bh * 2 + s) * 32 + t], ksacc);
}

// --------------------- out = q + sum_s (q . kv_s) / (q . ksum_s) per token
__global__ __launch_bounds__(256) void attn_out_kernel(
    const bf16_t* __restrict__ q, const float* __restrict__ kv,
    const float* __restrict__ ksum, bf16_t* __restrict__ outb) {
  __shared__ float qs[CC];
  __shared__ float ksums[2 * HH * DD];  // 512
  int row = blockIdx.x;      // b*N + n
  int b = row / NN;
  int c = threadIdx.x;
  qs[c] = (float)q[(size_t)row * CC + c];
  ksums[c] = ksum[(size_t)b * 512 + c];
  ksums[c + 256] = ksum[(size_t)b * 512 + c + 256];
  __syncthreads();
  int h = c >> 5, e = c & 31;
  float acc = qs[c];
#pragma unroll
  for (int s = 0; s < 2; ++s) {
    const float* kvp = kv + ((size_t)(b * HH + h) * 2 + s) * 1024;
    const float* ksp = ksums + h * 64 + s * 32;
    float num = 0.f, den = 0.f;
#pragma unroll
    for (int d = 0; d < DD; ++d) {
      float qd = qs[h * DD + d];
      num += qd * kvp[(size_t)d * 32 + e];
      den += qd * ksp[d];
    }
    acc += num / den;
  }
  outb[(size_t)row * CC + c] = (bf16_t)acc;
}

// ---------------------------------------------------------------- launcher
extern "C" void kernel_launch(void* const* d_in, const int* in_sizes, int n_in,
                              void* d_out, int out_size, void* d_ws, size_t ws_size,
                              hipStream_t stream) {
  const float* input = (const float*)d_in[0];
  const float* fx    = (const float*)d_in[1];
  const float* inF   = (const float*)d_in[2];
  const float* ln1g  = (const float*)d_in[3];
  const float* ln1b  = (const float*)d_in[4];
  const float* ln2g  = (const float*)d_in[5];
  const float* ln2b  = (const float*)d_in[6];
  const float* ln3g  = (const float*)d_in[7];
  const float* ln3b  = (const float*)d_in[8];
  const float* Ws[8] = {(const float*)d_in[9],  (const float*)d_in[11],
                        (const float*)d_in[13], (const float*)d_in[15],
                        (const float*)d_in[17], (const float*)d_in[19],
                        (const float*)d_in[21], (const float*)d_in[23]};
  const float* bq  = (const float*)d_in[10];
  const float* bk0 = (const float*)d_in[12];
  const float* bv0 = (const float*)d_in[14];
  const float* bk1 = (const float*)d_in[16];
  const float* bv1 = (const float*)d_in[18];
  const float* bo  = (const float*)d_in[20];
  const float* bm1 = (const float*)d_in[22];
  const float* bm2 = (const float*)d_in[24];
  const float* mu  = (const float*)d_in[25];

  const size_t NCf = (size_t)BB * NN * CC;
  float* xOut  = (float*)d_out;        // final x  [B,N,C]
  float* fxOut = (float*)d_out + NCf;  // fx_new   [B,N,C]
  float* x1    = fxOut;                // pre-MLP residual aliases fx slot

  char* ws = (char*)d_ws;
  const size_t SZB = NCf * sizeof(bf16_t);   // 16 MB
  bf16_t* fxT  = (bf16_t*)(ws + 0 * SZB);
  bf16_t* xn   = (bf16_t*)(ws + 1 * SZB);
  bf16_t* fn   = (bf16_t*)(ws + 2 * SZB);
  bf16_t* qb   = (bf16_t*)(ws + 3 * SZB);
  bf16_t* k0b  = (bf16_t*)(ws + 4 * SZB);
  bf16_t* v0b  = (bf16_t*)(ws + 5 * SZB);
  bf16_t* k1b  = (bf16_t*)(ws + 6 * SZB);
  bf16_t* v1b  = (bf16_t*)(ws + 7 * SZB);
  bf16_t* outb = (bf16_t*)(ws + 8 * SZB);
  bf16_t* yb   = xn;    // reuse (xn dead after v1 GEMM)
  bf16_t* hb   = fn;    // reuse (fn dead after v0 GEMM)
  bf16_t* xmub = qb;    // reuse (q dead after attn combine)
  bf16_t* xTb  = k0b;   // reuse (k0 dead after kv)
  char* tail = ws + 9 * SZB;
  bf16_t* xtfxT = (bf16_t*)tail;                               // B*65536 bf16
  float*  kvb   = (float*)(tail + (size_t)BB * 65536 * 2);     // B*H*2*1024 f32
  float*  ksumb = kvb + (size_t)BB * HH * 2 * 1024;            // B*H*2*32 f32
  bf16_t* wT    = (bf16_t*)(ksumb + (size_t)BB * HH * 2 * 32); // 8 * 65536 bf16

  dim3 b256(256), b32x8(32, 8);

  // weights -> bf16 [n][k]
  for (int i = 0; i < 8; ++i)
    transpose_bf_kernel<<<dim3(8, 8, 1), b32x8, 0, stream>>>(
        Ws[i], wT + (size_t)i * 65536, CC, CC);
  // fx -> bf16 [B,C,N]
  transpose_bf_kernel<<<dim3(NN / 32, CC / 32, BB), b32x8, 0, stream>>>(fx, fxT, NN, CC);
  // PE + LN1/LN2
  prep_kernel<<<dim3(BB * NN), b256, 0, stream>>>(input, inF, ln1g, ln1b, ln2g, ln2b, xn, fn);

  // q/k/v projections (softmax-over-head fused for q,k)
  gemm256<1, CC><<<dim3(256, 1), b256, 0, stream>>>(fn, wT + 0 * 65536, bq,  nullptr, nullptr, nullptr, nullptr, qb,  BB * NN, 0, 0, 0, 0);
  gemm256<1, CC><<<dim3(256, 1), b256, 0, stream>>>(fn, wT + 1 * 65536, bk0, nullptr, nullptr, nullptr, nullptr, k0b, BB * NN, 0, 0, 0, 0);
  gemm256<0, CC><<<dim3(256, 1), b256, 0, stream>>>(fn, wT + 2 * 65536, bv0, nullptr, nullptr, nullptr, nullptr, v0b, BB * NN, 0, 0, 0, 0);
  gemm256<1, CC><<<dim3(256, 1), b256, 0, stream>>>(xn, wT + 3 * 65536, bk1, nullptr, nullptr, nullptr, nullptr, k1b, BB * NN, 0, 0, 0, 0);
  gemm256<0, CC><<<dim3(256, 1), b256, 0, stream>>>(xn, wT + 4 * 65536, bv1, nullptr, nullptr, nullptr, nullptr, v1b, BB * NN, 0, 0, 0, 0);

  // kv / ksum accumulation
  int zn = BB * HH * 2 * 1024 + BB * HH * 2 * 32;
  zero_kernel<<<dim3((zn + 255) / 256), b256, 0, stream>>>(kvb, zn);
  kv_kernel<<<dim3(NN / 256, BB * HH), b256, 0, stream>>>(k0b, v0b, kvb, ksumb, 0);
  kv_kernel<<<dim3(NN / 256, BB * HH), b256, 0, stream>>>(k1b, v1b, kvb, ksumb, 1);

  // linear-attention combine
  attn_out_kernel<<<dim3(BB * NN), b256, 0, stream>>>(qb, kvb, ksumb, outb);

  // out@Wo + bo + input -> x1 (f32) and LN3(x1) -> yb
  gemm256<2, CC><<<dim3(256, 1), b256, 0, stream>>>(outb, wT + 5 * 65536, bo, input, ln3g, ln3b, x1, yb, BB * NN, 0, 0, 0, 0);
  // GELU(y@Wm1 + bm1) -> hb
  gemm256<3, CC><<<dim3(256, 1), b256, 0, stream>>>(yb, wT + 6 * 65536, bm1, nullptr, nullptr, nullptr, nullptr, hb, BB * NN, 0, 0, 0, 0);
  // x = x1 + h@Wm2 + bm2 -> xOut (f32) and x*softplus(mu) -> xmub
  gemm256<4, CC><<<dim3(256, 1), b256, 0, stream>>>(hb, wT + 7 * 65536, bm2, x1, mu, nullptr, xOut, xmub, BB * NN, 0, 0, 0, 0);

  // x -> bf16 [B,C,N]
  transpose_bf_kernel<<<dim3(NN / 32, CC / 32, BB), b32x8, 0, stream>>>(xOut, xTb, NN, CC);
  // xtfx^T[b][f][c] = sum_n x[b,n,c] fx[b,n,f]  (K = N = 8192, batched)
  gemm256<5, NN><<<dim3(2, BB), b256, 0, stream>>>(xTb, fxT, nullptr, nullptr, nullptr, nullptr,
                                                   nullptr, xtfxT, CC,
                                                   (long)CC * NN, (long)CC * NN, 0, 65536);
  // fx_new = LeakyReLU( xmu @ xtfx ), per batch
  gemm256<6, CC><<<dim3(NN / 128, BB), b256, 0, stream>>>(xmub, xtfxT, nullptr, nullptr, nullptr, nullptr,
                                                          fxOut, nullptr, NN,
                                                          (long)NN * CC, 65536, (long)NN * CC, 0);
}
